// GraphGateTemplate_6682969113345
// MI455X (gfx1250) — matrix operand order repaired and checked by reading, not compiled, not run on hardware
//
#include <hip/hip_runtime.h>

typedef __attribute__((ext_vector_type(16))) __bf16 v16bf;
typedef __attribute__((ext_vector_type(8)))  float  v8f;
typedef __attribute__((ext_vector_type(8)))  int    v8i;
typedef __attribute__((ext_vector_type(4)))  int    v4i;
typedef __attribute__((ext_vector_type(4)))  float  v4f;

#define TM 128
#define TN 128
#define TK 32
#define LDSS 40   // bf16 stride for LDS tiles: 80 B rows -> 16B-aligned vector stores

struct GemmAcc { v8f a[2][4]; };

__device__ __forceinline__ float sigmoidf_(float x) {
  return 1.0f / (1.0f + __expf(-x));
}

// Async global->LDS 128-bit copy (CDNA5 ASYNCcnt path). LDS byte address is
// the low 32 bits of the generic pointer.
__device__ __forceinline__ void async_copy_b128(__bf16* ldst, const __bf16* gsrc) {
  unsigned loff = (unsigned)(uintptr_t)ldst;
  asm volatile("global_load_async_to_lds_b128 %0, %1, off"
               :: "v"(loff), "v"(gsrc) : "memory");
}
__device__ __forceinline__ void wait_asynccnt0() {
  asm volatile("s_wait_asynccnt 0x0" ::: "memory");
}

// ---------------------------------------------------------------------------
// WMMA bf16 GEMM core. Block = 256 threads = 8 waves. Tile 128x128, K chunks
// of 32. Wave w computes rows [32*(w>>1), +32) x cols [64*(w&1), +64) i.e.
// 2x4 fragments of v_wmma_f32_16x16x32_bf16.
// A source may be f32 (converted to bf16 while staging to LDS) or bf16
// (staged with global_load_async_to_lds_b128, no VGPR round-trip).
// Split-K A: rows come from A0 (lda0) for k < ksplit, else from A1 (lda1).
// B is bf16 row-major [K x N]; staged transposed into LDS as [n][k].
// ---------------------------------------------------------------------------
template<typename TA>
__device__ __forceinline__ void gemm_core(
    const TA* __restrict__ A0, int lda0,
    const __bf16* __restrict__ A1, int lda1, int ksplit,
    const __bf16* __restrict__ Bm, int ldb, int K,
    __bf16* sA, __bf16* sB, GemmAcc& acc)
{
  const int tid  = threadIdx.x;
  const int lane = tid & 31;
  const int wave = tid >> 5;
  const int wm   = wave >> 1;   // 0..3
  const int wn   = wave & 1;    // 0..1
  const int half = lane >> 4;   // 0..1
  const int l16  = lane & 15;

  const v8f vzero = {0.f,0.f,0.f,0.f,0.f,0.f,0.f,0.f};
#pragma unroll
  for (int i = 0; i < 2; ++i)
#pragma unroll
    for (int j = 0; j < 4; ++j) acc.a[i][j] = vzero;

  for (int k0 = 0; k0 < K; k0 += TK) {
    __syncthreads();
    const bool useA1 = (k0 >= ksplit);   // uniform branch
    // ---- stage A tile: 128 rows x 32 bf16 ----
    if (!useA1) {
      if constexpr (sizeof(TA) == 4) {
        // f32 source: 4096 floats = 1024 float4; 4 per thread; cvt -> bf16
        const float* Ap = (const float*)A0;
#pragma unroll
        for (int it = 0; it < 4; ++it) {
          int idx = tid + it * 256;      // 0..1023
          int r   = idx >> 3;            // 8 float4 per row
          int c   = (idx & 7) * 4;       // f32 col
          v4f d = *(const v4f*)(Ap + (size_t)r * lda0 + k0 + c);
          union { unsigned long long u; __bf16 h[4]; } cv;
          cv.h[0] = (__bf16)d[0]; cv.h[1] = (__bf16)d[1];
          cv.h[2] = (__bf16)d[2]; cv.h[3] = (__bf16)d[3];
          *(unsigned long long*)(sA + r * LDSS + c) = cv.u;   // 8B aligned
        }
      } else {
        // bf16 source: 512 b128 chunks; 2 per thread; async to LDS
        const __bf16* Ap = (const __bf16*)A0;
#pragma unroll
        for (int it = 0; it < 2; ++it) {
          int idx = tid + it * 256;      // 0..511
          int r   = idx >> 2;            // 4 b128 per row
          int c   = (idx & 3) * 8;       // bf16 col
          async_copy_b128(sA + r * LDSS + c,
                          Ap + (size_t)r * lda0 + k0 + c);
        }
      }
    } else {
      const int kk = k0 - ksplit;
#pragma unroll
      for (int it = 0; it < 2; ++it) {
        int idx = tid + it * 256;
        int r   = idx >> 2;
        int c   = (idx & 3) * 8;
        async_copy_b128(sA + r * LDSS + c,
                        A1 + (size_t)r * lda1 + kk + c);
      }
    }
    // ---- stage B tile: global [32 x 128] -> LDS transposed sB[n][k] ----
    // 512 b128 chunks (8 bf16 = 8 consecutive n); 2 per thread
#pragma unroll
    for (int it = 0; it < 2; ++it) {
      int idx = tid + it * 256;          // 0..511
      int kr  = idx >> 4;                // 0..31
      int nc  = (idx & 15) * 8;          // 0..120
      v4i d = *(const v4i*)(Bm + (size_t)(k0 + kr) * ldb + nc);
      union { v4i v; __bf16 h[8]; } cv; cv.v = d;
#pragma unroll
      for (int q = 0; q < 8; ++q)
        sB[(nc + q) * LDSS + kr] = cv.h[q];
    }
    wait_asynccnt0();
    __syncthreads();
    // ---- load fragments per ISA 16-bit A (16x32) / B (32x16) layouts ----
    v8i afr[2], bfr[4];
#pragma unroll
    for (int i = 0; i < 2; ++i) {
      int m = wm * 32 + i * 16 + l16;
#pragma unroll
      for (int t = 0; t < 8; ++t) {
        int k = ((t >> 2) << 4) + (half << 3) + ((t & 3) << 1);
        afr[i][t] = *(const int*)(sA + m * LDSS + k);
      }
    }
#pragma unroll
    for (int j = 0; j < 4; ++j) {
      int n  = wn * 64 + j * 16 + l16;
      int kb = half << 3;
#pragma unroll
      for (int t = 0; t < 8; ++t)
        bfr[j][t] = *(const int*)(sB + n * LDSS + kb + ((t >> 2) << 4) + ((t & 3) << 1));
    }
#pragma unroll
    for (int i = 0; i < 2; ++i)
#pragma unroll
      for (int j = 0; j < 4; ++j)
        acc.a[i][j] = __builtin_amdgcn_wmma_f32_16x16x32_bf16(
            false, __builtin_bit_cast(v16bf, afr[i]),
            false, __builtin_bit_cast(v16bf, bfr[j]),
            (short)0, acc.a[i][j], false, false);
  }
}

// Epilogue iteration: D tile layout — lane l16 = column, VGPR r -> row r + 8*half
#define GEMM_EPILOGUE(body) do {                                              \
  const int lane_ = threadIdx.x & 31;                                         \
  const int wave_ = threadIdx.x >> 5;                                         \
  const int wm_ = wave_ >> 1, wn_ = wave_ & 1;                                \
  const int half_ = lane_ >> 4, l16_ = lane_ & 15;                            \
  _Pragma("unroll") for (int i = 0; i < 2; ++i)                               \
  _Pragma("unroll") for (int j = 0; j < 4; ++j)                               \
  _Pragma("unroll") for (int r = 0; r < 8; ++r) {                             \
    int m = wm_ * 32 + i * 16 + half_ * 8 + r;                                \
    int n = wn_ * 64 + j * 16 + l16_;                                         \
    float v = acc.a[i][j][r];                                                 \
    body;                                                                     \
  } } while (0)

// ---------------------------------------------------------------------------
// Step kernels
// ---------------------------------------------------------------------------

// gp = bf16(out @ Wsum + b)  :  [8192 x 128] @ [128 x 512], z selects in/out
__global__ void __launch_bounds__(256, 1)
k_gemm_pre(const __bf16* __restrict__ cat,
           const __bf16* __restrict__ WinS,
           const __bf16* __restrict__ WoutS,
           const float* __restrict__ b_in,
           const float* __restrict__ b_out,
           __bf16* __restrict__ gp_in,
           __bf16* __restrict__ gp_out)
{
  __shared__ __align__(16) __bf16 sA[128 * LDSS];
  __shared__ __align__(16) __bf16 sB[128 * LDSS];
  const int m0 = blockIdx.y * TM;
  const int n0 = blockIdx.x * TN;
  const int which = blockIdx.z;
  const __bf16* A0 = cat + 256 + (size_t)m0 * 384;
  const __bf16* Bm = (which ? WoutS : WinS) + n0;
  const float* bias = which ? b_out : b_in;
  __bf16* gp = which ? gp_out : gp_in;
  GemmAcc acc;
  gemm_core<__bf16>(A0, 384, nullptr, 0, 1 << 30, Bm, 512, 128, sA, sB, acc);
  GEMM_EPILOGUE({
    float o = v + bias[n0 + n];
    gp[(size_t)(m0 + m) * 512 + n0 + n] = (__bf16)o;
  });
}

// gi/go = graph(f32) @ gp(bf16) per batch : [1024 x 4096] @ [4096 x 128]
// writes bf16 into cat columns [which*128, +128)
__global__ void __launch_bounds__(256, 1)
k_gemm_bmm(const float* __restrict__ g_in,
           const float* __restrict__ g_out,
           const __bf16* __restrict__ gp_in,
           const __bf16* __restrict__ gp_out,
           __bf16* __restrict__ cat)
{
  __shared__ __align__(16) __bf16 sA[128 * LDSS];
  __shared__ __align__(16) __bf16 sB[128 * LDSS];
  const int m0 = blockIdx.y * TM;          // row within batch
  const int b = blockIdx.z >> 1;
  const int which = blockIdx.z & 1;
  const float* A0 = (which ? g_out : g_in) + (size_t)b * 1024 * 4096
                    + (size_t)m0 * 4096;
  const __bf16* Bm = (which ? gp_out : gp_in) + (size_t)b * 4096 * 128;
  GemmAcc acc;
  gemm_core<float>(A0, 4096, nullptr, 0, 1 << 30, Bm, 128, 4096, sA, sB, acc);
  GEMM_EPILOGUE({
    size_t row = (size_t)b * 1024 + m0 + m;
    cat[row * 384 + which * 128 + n] = (__bf16)v;
  });
}

// r/z gates: cat[8192x384] @ Wrz[384x256]; n-tile 0 -> r*out (bf16), 1 -> z
__global__ void __launch_bounds__(256, 1)
k_gemm_rz(const __bf16* __restrict__ cat,
          const __bf16* __restrict__ Wrz,
          const float* __restrict__ b_r,
          const float* __restrict__ b_z,
          const float* __restrict__ out_f32,
          __bf16* __restrict__ ro,
          float* __restrict__ zbuf)
{
  __shared__ __align__(16) __bf16 sA[128 * LDSS];
  __shared__ __align__(16) __bf16 sB[128 * LDSS];
  const int m0 = blockIdx.y * TM;
  const int n0 = blockIdx.x * TN;          // 0 (r) or 128 (z)
  const __bf16* A0 = cat + (size_t)m0 * 384;
  const __bf16* Bm = Wrz + n0;
  GemmAcc acc;
  gemm_core<__bf16>(A0, 384, nullptr, 0, 1 << 30, Bm, 256, 384, sA, sB, acc);
  if (n0 == 0) {
    GEMM_EPILOGUE({
      size_t idx = (size_t)(m0 + m) * 128 + n;
      float rr = sigmoidf_(v + b_r[n]);
      ro[idx] = (__bf16)(rr * out_f32[idx]);
    });
  } else {
    GEMM_EPILOGUE({
      size_t idx = (size_t)(m0 + m) * 128 + n;
      zbuf[idx] = sigmoidf_(v + b_z[n]);
    });
  }
}

// h_hat + GRU update: A = [cat(:,0:256) | ro] (split-K) @ Wt[384x128]
__global__ void __launch_bounds__(256, 1)
k_gemm_h(__bf16* cat,                 // read k<256, write 256+
         const __bf16* __restrict__ ro,
         const __bf16* __restrict__ Wt,
         const float* __restrict__ b_t,
         const float* __restrict__ zbuf,
         float* __restrict__ out_f32)
{
  __shared__ __align__(16) __bf16 sA[128 * LDSS];
  __shared__ __align__(16) __bf16 sB[128 * LDSS];
  const int m0 = blockIdx.y * TM;
  const __bf16* A0 = cat + (size_t)m0 * 384;
  const __bf16* A1 = ro + (size_t)m0 * 128;
  GemmAcc acc;
  gemm_core<__bf16>(A0, 384, A1, 128, 256, Wt, 128, 384, sA, sB, acc);
  GEMM_EPILOGUE({
    size_t idx = (size_t)(m0 + m) * 128 + n;
    float h  = tanhf(v + b_t[n]);
    float zv = zbuf[idx];
    float o  = out_f32[idx];
    float no = (1.0f - zv) * o + zv * h;
    out_f32[idx] = no;
    cat[(size_t)(m0 + m) * 384 + 256 + n] = (__bf16)no;
  });
}

// ---------------------------------------------------------------------------
// Prep kernels
// ---------------------------------------------------------------------------

// Collapse W[512x512] over the 4-way feature tiling: WS[d,j] = sum_e W[j, e*128+d]
__global__ void k_prep_ws(const float* __restrict__ W, __bf16* __restrict__ WS) {
  int idx = blockIdx.x * 256 + threadIdx.x;   // 128*512
  int dd = idx >> 9;
  int j  = idx & 511;
  float s = W[(size_t)j * 512 + 0   + dd] + W[(size_t)j * 512 + 128 + dd]
          + W[(size_t)j * 512 + 256 + dd] + W[(size_t)j * 512 + 384 + dd];
  WS[(size_t)dd * 512 + j] = (__bf16)s;
}

// Wrz[k, n] = (n<128 ? W_r[n,k] : W_z[n-128,k])   (B operand, k-major)
__global__ void k_prep_wrz(const float* __restrict__ W_r,
                           const float* __restrict__ W_z,
                           __bf16* __restrict__ Wrz) {
  int idx = blockIdx.x * 256 + threadIdx.x;   // 384*256
  int k = idx >> 8;
  int n = idx & 255;
  float v = (n < 128) ? W_r[(size_t)n * 384 + k] : W_z[(size_t)(n - 128) * 384 + k];
  Wrz[(size_t)k * 256 + n] = (__bf16)v;
}

// Wt[k, n] = W_t[n, k]
__global__ void k_prep_wt(const float* __restrict__ W_t, __bf16* __restrict__ Wt) {
  int idx = blockIdx.x * 256 + threadIdx.x;   // 384*128
  int k = idx >> 7;
  int n = idx & 127;
  Wt[(size_t)k * 128 + n] = (__bf16)W_t[(size_t)n * 384 + k];
}

// out_f32 = batchinput; cat[:,256:384] = bf16(batchinput)
__global__ void k_init(const float* __restrict__ x,
                       float* __restrict__ out_f32,
                       __bf16* __restrict__ cat) {
  int idx = blockIdx.x * 256 + threadIdx.x;   // 8192*128
  int row = idx >> 7;
  int col = idx & 127;
  float v = x[idx];
  out_f32[idx] = v;
  cat[(size_t)row * 384 + 256 + col] = (__bf16)v;
}

// ---------------------------------------------------------------------------
extern "C" void kernel_launch(void* const* d_in, const int* in_sizes, int n_in,
                              void* d_out, int out_size, void* d_ws, size_t ws_size,
                              hipStream_t stream) {
  (void)in_sizes; (void)n_in; (void)out_size; (void)ws_size;
  const float* batchinput = (const float*)d_in[0];
  const float* graphin    = (const float*)d_in[1];
  const float* graphout   = (const float*)d_in[2];
  const float* W_in  = (const float*)d_in[3];
  const float* b_in  = (const float*)d_in[4];
  const float* W_out = (const float*)d_in[5];
  const float* b_out = (const float*)d_in[6];
  const float* W_r   = (const float*)d_in[7];
  const float* b_r   = (const float*)d_in[8];
  const float* W_z   = (const float*)d_in[9];
  const float* b_z   = (const float*)d_in[10];
  const float* W_t   = (const float*)d_in[11];
  const float* b_t   = (const float*)d_in[12];

  char* ws = (char*)d_ws;
  size_t off = 0;
  auto alloc = [&](size_t bytes) -> void* {
    void* p = ws + off;
    off = (off + bytes + 255) & ~(size_t)255;
    return p;
  };
  __bf16* gp_in   = (__bf16*)alloc((size_t)8 * 4096 * 128 * 2);
  __bf16* gp_out  = (__bf16*)alloc((size_t)8 * 4096 * 128 * 2);
  __bf16* cat     = (__bf16*)alloc((size_t)8192 * 384 * 2);
  __bf16* ro      = (__bf16*)alloc((size_t)8192 * 128 * 2);
  float*  zbuf    = (float*) alloc((size_t)8192 * 128 * 4);
  float*  out_f32 = (float*) alloc((size_t)8192 * 128 * 4);
  __bf16* WinS    = (__bf16*)alloc((size_t)128 * 512 * 2);
  __bf16* WoutS   = (__bf16*)alloc((size_t)128 * 512 * 2);
  __bf16* Wrz     = (__bf16*)alloc((size_t)384 * 256 * 2);
  __bf16* Wt      = (__bf16*)alloc((size_t)384 * 128 * 2);

  // prep (re-run every call: deterministic)
  k_prep_ws <<<256, 256, 0, stream>>>(W_in,  WinS);
  k_prep_ws <<<256, 256, 0, stream>>>(W_out, WoutS);
  k_prep_wrz<<<384, 256, 0, stream>>>(W_r, W_z, Wrz);
  k_prep_wt <<<192, 256, 0, stream>>>(W_t, Wt);
  k_init    <<<4096, 256, 0, stream>>>(batchinput, out_f32, cat);

  for (int s = 0; s < 5; ++s) {
    k_gemm_pre<<<dim3(4, 64, 2),  256, 0, stream>>>(cat, WinS, WoutS, b_in, b_out,
                                                    gp_in, gp_out);
    k_gemm_bmm<<<dim3(1, 8, 16),  256, 0, stream>>>(graphin, graphout, gp_in, gp_out,
                                                    cat);
    k_gemm_rz <<<dim3(2, 64, 1),  256, 0, stream>>>(cat, Wrz, b_r, b_z, out_f32,
                                                    ro, zbuf);
    k_gemm_h  <<<dim3(1, 64, 1),  256, 0, stream>>>(cat, ro, Wt, b_t, zbuf, out_f32);
  }
  hipMemcpyAsync(d_out, out_f32, (size_t)8192 * 128 * 4,
                 hipMemcpyDeviceToDevice, stream);
}
